// ResidualGCNPaper_618475290794
// MI455X (gfx1250) — matrix operand order repaired and checked by reading, not compiled, not run on hardware
//
#include <hip/hip_runtime.h>
#include <hip/hip_bf16.h>
#include <math.h>

typedef __attribute__((ext_vector_type(16))) __bf16 v16bf;
typedef __attribute__((ext_vector_type(8)))  float  v8f;

#define IN_DIM 256
#define HID    128
#define OUTF   64

// ---------------------------------------------------------------------------
// utility kernels
// ---------------------------------------------------------------------------
__global__ void rg_zero_f32(float* p, int n) {
    int i = blockIdx.x * blockDim.x + threadIdx.x;
    if (i < n) p[i] = 0.0f;
}

__global__ void rg_count_deg(const long long* __restrict__ ei, float* __restrict__ deg,
                             long long E) {
    long long e = (long long)blockIdx.x * blockDim.x + threadIdx.x;
    if (e >= E) return;
    int dst = (int)ei[E + e];
    unsafeAtomicAdd(&deg[dst], 1.0f);
}

__global__ void rg_finalize_dinv(float* __restrict__ dinv, int n) {
    int i = blockIdx.x * blockDim.x + threadIdx.x;
    if (i < n) dinv[i] = rsqrtf(dinv[i] + 1.0f);  // deg includes +1 self loop
}

// Pre-swizzle W (row-major [K, Ncols], f32) into WMMA bf16 B-fragments.
// Fragment layout per (ntile, kchunk): 32 lanes x 16 bf16.
//   lanes 0-15 : col = nt*16 + lane,      elem j -> K = kc*32 + (j&7) + (j>>3)*16
//   lanes 16-31: col = nt*16 + lane - 16, elem j -> K = kc*32 + 8 + (j&7) + (j>>3)*16
__global__ void rg_swizzle_w(const float* __restrict__ W, __bf16* __restrict__ frag,
                             int K, int Ncols) {
    int tid = blockIdx.x * blockDim.x + threadIdx.x;
    int total = K * Ncols;
    if (tid >= total) return;
    int j    = tid & 15;
    int lane = (tid >> 4) & 31;
    int KC   = K >> 5;
    int kc   = (tid >> 9) % KC;
    int nt   = tid / (512 * KC);
    int col  = nt * 16 + (lane & 15);
    int kk   = kc * 32 + (lane >> 4) * 8 + (j & 7) + (j >> 3) * 16;
    frag[tid] = (__bf16)W[kk * Ncols + col];
}

// ---------------------------------------------------------------------------
// Multi-N WMMA GEMM: one wave per 16-row M-tile, NT column tiles accumulated
// in registers (A fragment loaded once per k-chunk, reused NT times).
// Column tiles [0, NT0) -> C0 (Nc0 cols); [NT0, NT) -> C1 (Nc1 cols).
// K multiple of 32, M multiple of 16.
// ---------------------------------------------------------------------------
template <int NT, int NT0>
__global__ void rg_gemm_wmma_multiN(const float* __restrict__ A,
                                    const __bf16* __restrict__ wfrag,
                                    float* __restrict__ C0, int Nc0,
                                    float* __restrict__ C1, int Nc1,
                                    int M, int K) {
    int wid  = blockIdx.x * (blockDim.x / 32) + (threadIdx.x >> 5);
    int lane = threadIdx.x & 31;
    int tiles_m = M >> 4;
    if (wid >= tiles_m) return;

    int m0   = wid << 4;
    int half = lane >> 4;            // 0 or 1
    int r    = lane & 15;
    const float* arow = A + (size_t)(m0 + r) * K;

    int KC = K >> 5;
    const __bf16* wl = wfrag + (size_t)lane * 16;   // lane offset within a fragment

    v8f acc[NT];
    #pragma unroll
    for (int nt = 0; nt < NT; ++nt) acc[nt] = (v8f){};

    for (int kc = 0; kc < KC; ++kc) {
        // ---- A fragment (16-bit A 16x32 layout), loaded once per k-chunk ----
        int kbase = kc * 32 + half * 8;
        const float4* p0 = reinterpret_cast<const float4*>(arow + kbase);
        float4 f0 = p0[0];
        float4 f1 = p0[1];
        const float4* p1 = reinterpret_cast<const float4*>(arow + kbase + 16);
        float4 f2 = p1[0];
        float4 f3 = p1[1];
        v16bf a;
        a[0] = (__bf16)f0.x;  a[1] = (__bf16)f0.y;  a[2]  = (__bf16)f0.z;  a[3]  = (__bf16)f0.w;
        a[4] = (__bf16)f1.x;  a[5] = (__bf16)f1.y;  a[6]  = (__bf16)f1.z;  a[7]  = (__bf16)f1.w;
        a[8] = (__bf16)f2.x;  a[9] = (__bf16)f2.y;  a[10] = (__bf16)f2.z;  a[11] = (__bf16)f2.w;
        a[12] = (__bf16)f3.x; a[13] = (__bf16)f3.y; a[14] = (__bf16)f3.z;  a[15] = (__bf16)f3.w;

        // ---- NT x (32B B-fragment load + WMMA), A reused ----
        #pragma unroll
        for (int nt = 0; nt < NT; ++nt) {
            v16bf b = *reinterpret_cast<const v16bf*>(
                wl + ((size_t)(nt * KC + kc) * 32) * 16);
            acc[nt] = __builtin_amdgcn_wmma_f32_16x16x32_bf16(
                false, a, false, b, (short)0, acc[nt], false, false);
        }
    }

    // ---- store (VGPR i: lanes 0-15 -> M=i, lanes 16-31 -> M=8+i) ----
    #pragma unroll
    for (int nt = 0; nt < NT; ++nt) {
        float* C  = (nt < NT0) ? C0 : C1;
        int    nc = (nt < NT0) ? Nc0 : Nc1;
        int    n0 = (nt < NT0) ? nt * 16 : (nt - NT0) * 16;
        #pragma unroll
        for (int i = 0; i < 8; ++i) {
            int row = m0 + half * 8 + i;
            C[(size_t)row * nc + n0 + r] = acc[nt][i];
        }
    }
}

// ---------------------------------------------------------------------------
// GCN aggregation pieces
// ---------------------------------------------------------------------------
// agg[i,f] = h[i,f] * dinv[i]^2   (self-loop term; also zero-initializes agg)
__global__ void rg_self_init(const float* __restrict__ h, const float* __restrict__ dinv,
                             float* __restrict__ agg, int n, int F) {
    int idx = blockIdx.x * blockDim.x + threadIdx.x;
    if (idx >= n * F) return;
    int row = idx / F;
    float d = dinv[row];
    agg[idx] = h[idx] * d * d;
}

// edge scatter: agg[dst] += h[src] * dinv[src]*dinv[dst]
// F/4 threads per edge, each thread does a float4 feature chunk.
__global__ void rg_edge_scatter(const long long* __restrict__ ei,
                                const float* __restrict__ dinv,
                                const float* __restrict__ h,
                                float* __restrict__ agg,
                                long long E, int F) {
    int tpe = F >> 2;
    long long tid = (long long)blockIdx.x * blockDim.x + threadIdx.x;
    long long e = tid / tpe;
    if (e >= E) return;
    int g = (int)(tid % tpe);
    int src = (int)ei[e];
    int dst = (int)ei[E + e];
    float coef = dinv[src] * dinv[dst];
    float4 hv = *reinterpret_cast<const float4*>(h + (size_t)src * F + g * 4);
    float* p = agg + (size_t)dst * F + g * 4;
    unsafeAtomicAdd(p + 0, hv.x * coef);
    unsafeAtomicAdd(p + 1, hv.y * coef);
    unsafeAtomicAdd(p + 2, hv.z * coef);
    unsafeAtomicAdd(p + 3, hv.w * coef);
}

// out[i,f] = relu(agg[i,f] + b[f])
__global__ void rg_bias_relu(const float* __restrict__ agg, const float* __restrict__ b,
                             float* __restrict__ out, int n, int F) {
    int idx = blockIdx.x * blockDim.x + threadIdx.x;
    if (idx >= n * F) return;
    out[idx] = fmaxf(agg[idx] + b[idx % F], 0.0f);
}

// ---------------------------------------------------------------------------
// final: z = agg2 + b2 + xWres; L2-normalize rows; softmax.  One wave per row.
// OUTF = 64 -> 2 features per lane (wave32).
// ---------------------------------------------------------------------------
__device__ inline float rg_wave_sum(float v) {
    #pragma unroll
    for (int m = 16; m >= 1; m >>= 1) v += __shfl_xor(v, m, 32);
    return v;
}
__device__ inline float rg_wave_max(float v) {
    #pragma unroll
    for (int m = 16; m >= 1; m >>= 1) v = fmaxf(v, __shfl_xor(v, m, 32));
    return v;
}

__global__ void rg_final_row(const float* __restrict__ agg2, const float* __restrict__ b2,
                             const float* __restrict__ xwres, float* __restrict__ out,
                             int n) {
    int row  = blockIdx.x * (blockDim.x / 32) + (threadIdx.x >> 5);
    int lane = threadIdx.x & 31;
    if (row >= n) return;
    size_t base = (size_t)row * OUTF;
    int f0 = lane, f1 = lane + 32;
    float v0 = agg2[base + f0] + b2[f0] + xwres[base + f0];
    float v1 = agg2[base + f1] + b2[f1] + xwres[base + f1];

    float ss = rg_wave_sum(v0 * v0 + v1 * v1);
    float denom = fmaxf(sqrtf(ss), 1e-12f);
    float z0 = v0 / denom, z1 = v1 / denom;

    out[base + f0] = z0;
    out[base + f1] = z1;

    float mx = rg_wave_max(fmaxf(z0, z1));
    float e0 = expf(z0 - mx), e1 = expf(z1 - mx);
    float s = rg_wave_sum(e0 + e1);
    float inv = 1.0f / s;
    size_t pbase = (size_t)n * OUTF + base;
    out[pbase + f0] = e0 * inv;
    out[pbase + f1] = e1 * inv;
}

// ---------------------------------------------------------------------------
// host launcher
// ---------------------------------------------------------------------------
extern "C" void kernel_launch(void* const* d_in, const int* in_sizes, int n_in,
                              void* d_out, int out_size, void* d_ws, size_t ws_size,
                              hipStream_t stream) {
    const float*     x    = (const float*)d_in[0];
    const long long* ei   = (const long long*)d_in[1];   // int64 edge_index [2,E]
    const float*     W1   = (const float*)d_in[2];
    const float*     b1   = (const float*)d_in[3];
    const float*     W2   = (const float*)d_in[4];
    const float*     b2   = (const float*)d_in[5];
    const float*     Wres = (const float*)d_in[6];
    float*           out  = (float*)d_out;

    const int       N = in_sizes[0] / IN_DIM;
    const long long E = in_sizes[1] / 2;

    // workspace carve-up (256B aligned)
    char*  ws = (char*)d_ws;
    size_t o  = 0;
    auto carve = [&](size_t bytes) { size_t r = o; o += (bytes + 255) & ~(size_t)255; return r; };
    float*  dinv  = (float*)(ws + carve((size_t)N * 4));
    float*  h1    = (float*)(ws + carve((size_t)N * HID * 4));   // h1 / h_relu
    float*  agg1  = (float*)(ws + carve((size_t)N * HID * 4));
    float*  h2    = (float*)(ws + carve((size_t)N * OUTF * 4));
    float*  agg2  = (float*)(ws + carve((size_t)N * OUTF * 4));
    float*  xwres = (float*)(ws + carve((size_t)N * OUTF * 4));
    // fused fragment buffer: [W1 ntiles 0..7][Wres ntiles 8..11], contiguous
    // (both 64KB/32KB regions are 256B multiples, so carve stays contiguous)
    __bf16* wcat  = (__bf16*)(ws + carve((size_t)IN_DIM * (HID + OUTF) * 2));
    __bf16* w1f   = wcat;
    __bf16* wrf   = wcat + (size_t)IN_DIM * HID;
    __bf16* w2f   = (__bf16*)(ws + carve((size_t)HID * OUTF * 2));
    (void)ws_size; (void)n_in; (void)out_size;

    const int T = 256;

    // degree -> dinv
    rg_zero_f32<<<(N + T - 1) / T, T, 0, stream>>>(dinv, N);
    rg_count_deg<<<(int)((E + T - 1) / T), T, 0, stream>>>(ei, dinv, E);
    rg_finalize_dinv<<<(N + T - 1) / T, T, 0, stream>>>(dinv, N);

    // weight fragment swizzles (tiny)
    rg_swizzle_w<<<(IN_DIM * HID + T - 1) / T, T, 0, stream>>>(W1, w1f, IN_DIM, HID);
    rg_swizzle_w<<<(IN_DIM * OUTF + T - 1) / T, T, 0, stream>>>(Wres, wrf, IN_DIM, OUTF);
    rg_swizzle_w<<<(HID * OUTF + T - 1) / T, T, 0, stream>>>(W2, w2f, HID, OUTF);

    // Fused GEMM: [h1 | xwres] = x @ [W1 | Wres]   (single pass over x)
    {
        int tiles_m = N / 16;                       // 3125
        rg_gemm_wmma_multiN<12, 8><<<(tiles_m + 7) / 8, T, 0, stream>>>(
            x, wcat, h1, HID, xwres, OUTF, N, IN_DIM);
    }

    // GCN layer 1 aggregation
    rg_self_init<<<(N * HID + T - 1) / T, T, 0, stream>>>(h1, dinv, agg1, N, HID);
    {
        long long threads = E * (HID / 4);
        rg_edge_scatter<<<(int)((threads + T - 1) / T), T, 0, stream>>>(ei, dinv, h1, agg1, E, HID);
    }
    rg_bias_relu<<<(N * HID + T - 1) / T, T, 0, stream>>>(agg1, b1, h1, N, HID);  // h1 <- relu

    // GEMM 2: h2 = h_relu @ W2  [N,128]x[128,64]
    {
        int tiles_m = N / 16;
        rg_gemm_wmma_multiN<4, 4><<<(tiles_m + 7) / 8, T, 0, stream>>>(
            h1, w2f, h2, OUTF, h2, OUTF, N, HID);
    }

    // GCN layer 2 aggregation
    rg_self_init<<<(N * OUTF + T - 1) / T, T, 0, stream>>>(h2, dinv, agg2, N, OUTF);
    {
        long long threads = E * (OUTF / 4);
        rg_edge_scatter<<<(int)((threads + T - 1) / T), T, 0, stream>>>(ei, dinv, h2, agg2, E, OUTF);
    }

    // z + normalize + softmax, one wave per row
    rg_final_row<<<(N + 7) / 8, T, 0, stream>>>(agg2, b2, xwres, out, N);
}
